// MultiHeadDotProductAttention_18270790877872
// MI455X (gfx1250) — matrix operand order repaired and checked by reading, not compiled, not run on hardware
//
#include <hip/hip_runtime.h>

typedef __attribute__((ext_vector_type(16))) _Float16 v16h;
typedef __attribute__((ext_vector_type(8)))  _Float16 v8h;
typedef __attribute__((ext_vector_type(8)))  float    v8f;
typedef __attribute__((ext_vector_type(4)))  float    v4f;
typedef __attribute__((ext_vector_type(2)))  unsigned v2u;

#define NH  16
#define HD  64
#define Bsz 8
#define Tsz 1024
#define Esz 1024
#define HT  (NH * HD)   // 1024

#define USE_ASYNC_COPY 1

// ---------------------------------------------------------------------------
// helpers
// ---------------------------------------------------------------------------
__device__ __forceinline__ unsigned pack2(float a, float b) {
  union { _Float16 h[2]; unsigned u; } x;
  x.h[0] = (_Float16)a; x.h[1] = (_Float16)b;
  return x.u;
}

// A fragment (16x32 MxK, f16): per-lane data is two contiguous 16B chunks.
__device__ __forceinline__ v16h load_a16(const _Float16* p, int lda) {
  const int lane = threadIdx.x & 31;
  const int m = lane & 15, hs = lane >> 4;
  const _Float16* q = p + m * lda + hs * 16;
  const v8h lo = *(const v8h*)(q);
  const v8h hi = *(const v8h*)(q + 8);
  v16h a;
#pragma unroll
  for (int e = 0; e < 8; ++e) { a[e] = lo[e]; a[8 + e] = hi[e]; }
  return a;
}

// B fragment from swizzled 512-half blob: one contiguous 32B read per lane.
__device__ __forceinline__ v16h load_b_swz(const _Float16* blob) {
  const int lane = threadIdx.x & 31;
  return *(const v16h*)(blob + lane * 16);
}

__device__ __forceinline__ v8f wmma_f16(v16h a, v16h b, v8f c) {
  return __builtin_amdgcn_wmma_f32_16x16x32_f16(false, a, false, b,
                                                (short)0, c, false, false);
}

#if USE_ASYNC_COPY
// CDNA5 async copy: 16B per lane, global -> LDS, tracked by ASYNCcnt.
__device__ __forceinline__ void async_copy_b128(const _Float16* gsrc,
                                                _Float16* ldst) {
  unsigned lds = (unsigned)(size_t)ldst;   // flat addr low 32 bits == LDS offset
  asm volatile("global_load_async_to_lds_b128 %0, %1, off"
               :: "v"(lds), "v"(gsrc) : "memory");
}
__device__ __forceinline__ void wait_async0() {
  asm volatile("s_wait_asynccnt 0" ::: "memory");
}
#endif

// ===========================================================================
// Kernel 1: fused QKV projection + RoPE + f16 pack.
//   grid = (BT/128, NH, 3)   z: 0=Q(rope), 1=K(rope,swizzled), 2=V(swizzled)
//   K/V are written directly in WMMA-B-fragment blob layout:
//   per (b,h): [kvblk(16)][kc(2)][nt(4)][lane(32)][i(16)]  (512-half blobs)
//     K blobs: k-dim = d, n-dim = t'      V blobs: k-dim = t', n-dim = d
// ===========================================================================
__global__ void qkv_rope_kernel(const float* __restrict__ Xq,
                                const float* __restrict__ Xkv,
                                const float* __restrict__ Wq,
                                const float* __restrict__ Wk,
                                const float* __restrict__ Wv,
                                _Float16* __restrict__ Qh,    // [B,H,T,D]
                                _Float16* __restrict__ Ksw,   // swizzled
                                _Float16* __restrict__ Vsw) { // swizzled
  __shared__ _Float16 sA[128 * 32];   // 8 KB
  __shared__ _Float16 sB[32 * 64];    // 4 KB, swizzled blobs
  __shared__ float    sO[128 * 64];   // 32 KB

  const int tid   = threadIdx.x;
  const int which = blockIdx.z;
  const int h     = blockIdx.y;
  const int gr0   = blockIdx.x * 128;

  const float* X = (which == 0) ? Xq : Xkv;
  const float* W = (which == 0) ? Wq : ((which == 1) ? Wk : Wv);

  const int wid = tid >> 5;
  const int wm  = wid & 3;
  const int wn  = wid >> 2;

  const v8f vzero = {};
  v8f acc[2][2];
  acc[0][0] = vzero; acc[0][1] = vzero; acc[1][0] = vzero; acc[1][1] = vzero;

  for (int kc = 0; kc < Esz / 32; ++kc) {
    // ---- stage A tile 128x32: float4 loads, packed b64 LDS stores
#pragma unroll
    for (int i = 0; i < 4; ++i) {
      const int g  = tid + i * 256;          // 1024 groups of 4 floats
      const int r  = g >> 3, c4 = g & 7;
      const v4f x = *(const v4f*)&X[(size_t)(gr0 + r) * Esz + kc * 32 + c4 * 4];
      v2u w; w.x = pack2(x.x, x.y); w.y = pack2(x.z, x.w);
      *(v2u*)&sA[r * 32 + c4 * 4] = w;
    }
    // ---- stage B tile 32x64 into swizzled blobs (pairs along k -> b32)
    {
      const int kp = tid >> 4, n4 = tid & 15;
      const int k0 = kp * 2, n0 = n4 * 4;
      const float* wr = &W[(size_t)(kc * 32 + k0) * HT + h * HD + n0];
      const v4f r0 = *(const v4f*)wr;
      const v4f r1 = *(const v4f*)(wr + HT);
#pragma unroll
      for (int j = 0; j < 4; ++j) {
        const int n   = n0 + j;
        const int off = (n >> 4) * 512 + (((k0 >> 4) << 4) + (n & 15)) * 16 + (k0 & 15);
        *(unsigned*)&sB[off] = pack2(r0[j], r1[j]);
      }
    }
    __syncthreads();
#pragma unroll
    for (int ms = 0; ms < 2; ++ms) {
      const v16h a = load_a16(&sA[(wm * 32 + ms * 16) * 32], 32);
#pragma unroll
      for (int ns = 0; ns < 2; ++ns) {
        const v16h b = load_b_swz(&sB[(wn * 2 + ns) * 512]);
        acc[ms][ns] = wmma_f16(a, b, acc[ms][ns]);
      }
    }
    __syncthreads();
  }

  // ---- accumulators -> fp32 LDS tile (cross-lane RoPE pairing)
  {
    const int lane = tid & 31, hs = lane >> 4, ln = lane & 15;
#pragma unroll
    for (int ms = 0; ms < 2; ++ms)
#pragma unroll
      for (int ns = 0; ns < 2; ++ns)
#pragma unroll
        for (int r = 0; r < 8; ++r) {
          const int row = wm * 32 + ms * 16 + hs * 8 + r;
          const int col = wn * 32 + ns * 16 + ln;
          sO[row * 64 + col] = acc[ms][ns][r];
        }
  }
  __syncthreads();

  // ---- RoPE + pack f16 + store
  const float L2_10000 = 13.287712379549449f;   // log2(10000)
  const size_t bh_stride = (size_t)Tsz * HD;
#pragma unroll
  for (int i = 0; i < 16; ++i) {
    const int pidx = tid + i * 256;             // 4096 pairs
    const int row  = pidx >> 5;
    const int p    = pidx & 31;
    const int d0   = p * 2;
    const int gr   = gr0 + row;
    const int b    = gr >> 10;
    const int t    = gr & (Tsz - 1);
    const float x0 = sO[row * 64 + d0];
    const float x1 = sO[row * 64 + d0 + 1];
    float o0 = x0, o1 = x1;
    if (which < 2) {
      const float invf = exp2f(-(float)p * (L2_10000 / 32.0f));
      float s, c;
      sincosf((float)t * invf, &s, &c);
      o0 = x0 * c - x1 * s;
      o1 = x1 * c + x0 * s;
    }
    const size_t base = ((size_t)b * NH + h) * bh_stride;
    if (which == 0) {                       // Q: [B,H,T,D], paired store
      const size_t o = base + (size_t)t * HD + d0;
      *(unsigned*)&Qh[o] = pack2(o0, o1);
    } else if (which == 1) {                // K blobs: k=d, n=t'
      const int kvblk = t >> 6;
      const int nt    = (t >> 4) & 3;
      const int lane  = ((d0 >> 4) & 1) * 16 + (t & 15);
      const size_t o  = base + (size_t)(((kvblk * 2 + (d0 >> 5)) * 4 + nt) * 512)
                        + lane * 16 + (d0 & 15);
      *(unsigned*)&Ksw[o] = pack2(o0, o1); // i, i+1 contiguous
    } else {                                // V blobs: k=t', n=d
      const int kvblk = t >> 6;
      const int kc2   = (t >> 5) & 1;
      const int lb    = ((t >> 4) & 1) * 16;
      const int ii    = t & 15;
      const size_t o0i = base + (size_t)(((kvblk * 2 + kc2) * 4 + (d0 >> 4)) * 512)
                         + (lb + (d0 & 15)) * 16 + ii;
      Vsw[o0i]      = (_Float16)o0;
      Vsw[o0i + 16] = (_Float16)o1;       // next lane, same i
    }
  }
}

// ===========================================================================
// Kernel 2: flash attention per (b, h, 128-row q tile); kv tiles of 64.
//   K/V tiles arrive pre-swizzled -> staging is a pure contiguous copy,
//   done with CDNA5 async global->LDS b128 ops (ASYNCcnt).
// ===========================================================================
__global__ void attention_kernel(const _Float16* __restrict__ Qh,
                                 const _Float16* __restrict__ Ksw,
                                 const _Float16* __restrict__ Vsw,
                                 _Float16* __restrict__ Ctx) { // [B,T,HT]
  __shared__ _Float16 sK[4096];        // 8 KB : [kc(2)][nt(4)][512]
  __shared__ _Float16 sV[4096];        // 8 KB : [kc(2)][nt(4)][512]
  __shared__ _Float16 sSP[128 * 64];   // 16 KB: S then P (f16)
  __shared__ float sM[128], sL[128], sAl[128];

  const int tid = threadIdx.x;
  const int qt0 = blockIdx.x * 128;
  const int h   = blockIdx.y;
  const int b   = blockIdx.z;
  const int wid = tid >> 5, wm = wid & 3, wn = wid >> 2;
  const int lane = tid & 31, hs = lane >> 4, ln = lane & 15;

  const size_t bh = ((size_t)b * NH + h) * ((size_t)Tsz * HD);
  const _Float16* Qg = Qh + bh + (size_t)qt0 * HD;
  const _Float16* Kb = Ksw + bh;
  const _Float16* Vb = Vsw + bh;

  // Q fragments for the whole kernel (M=32 x K=64 per wave), b128 loads.
  v16h aq[2][2];
#pragma unroll
  for (int ms = 0; ms < 2; ++ms)
#pragma unroll
    for (int kc = 0; kc < 2; ++kc)
      aq[ms][kc] = load_a16(Qg + (wm * 32 + ms * 16) * HD + kc * 32, HD);

  const v8f vzero = {};
  v8f oacc[2][2];
  oacc[0][0] = vzero; oacc[0][1] = vzero; oacc[1][0] = vzero; oacc[1][1] = vzero;
  if (tid < 128) { sM[tid] = -1e30f; sL[tid] = 0.0f; }

  for (int kt = 0; kt < Tsz / 64; ++kt) {
    const _Float16* kg = Kb + kt * 4096;
    const _Float16* vg = Vb + kt * 4096;
#if USE_ASYNC_COPY
#pragma unroll
    for (int i = 0; i < 2; ++i) {
      const int g = tid + i * 256;           // 512 groups of 8 halves
      async_copy_b128(kg + g * 8, &sK[g * 8]);
      async_copy_b128(vg + g * 8, &sV[g * 8]);
    }
    wait_async0();
#else
#pragma unroll
    for (int i = 0; i < 2; ++i) {
      const int g = tid + i * 256;
      *(v8h*)&sK[g * 8] = *(const v8h*)(kg + g * 8);
      *(v8h*)&sV[g * 8] = *(const v8h*)(vg + g * 8);
    }
#endif
    __syncthreads();

    // ---- S = Q @ K^T (wave tile 32x32 of 128x64)
    v8f sacc[2][2];
    sacc[0][0] = vzero; sacc[0][1] = vzero; sacc[1][0] = vzero; sacc[1][1] = vzero;
#pragma unroll
    for (int kc = 0; kc < 2; ++kc) {
#pragma unroll
      for (int ns = 0; ns < 2; ++ns) {
        const v16h bf = load_b_swz(&sK[(kc * 4 + wn * 2 + ns) * 512]);
#pragma unroll
        for (int ms = 0; ms < 2; ++ms)
          sacc[ms][ns] = wmma_f16(aq[ms][kc], bf, sacc[ms][ns]);
      }
    }
#pragma unroll
    for (int ms = 0; ms < 2; ++ms)
#pragma unroll
      for (int ns = 0; ns < 2; ++ns)
#pragma unroll
        for (int r = 0; r < 8; ++r) {
          const int row = wm * 32 + ms * 16 + hs * 8 + r;
          const int col = wn * 32 + ns * 16 + ln;
          sSP[row * 64 + col] = (_Float16)(sacc[ms][ns][r] * 0.125f);
        }
    __syncthreads();

    // ---- online softmax, vectorized over v8h chunks, in-place S -> P
    if (tid < 128) {
      const int row = tid;
      v8h* rp = (v8h*)&sSP[row * 64];
      const float mold = sM[row];
      float mx = mold;
      v8h vv[8];
#pragma unroll
      for (int j = 0; j < 8; ++j) {
        vv[j] = rp[j];
#pragma unroll
        for (int e = 0; e < 8; ++e) mx = fmaxf(mx, (float)vv[j][e]);
      }
      const float alpha = __expf(mold - mx);
      float l = sL[row] * alpha;
#pragma unroll
      for (int j = 0; j < 8; ++j) {
        v8h pv;
#pragma unroll
        for (int e = 0; e < 8; ++e) {
          const float p = __expf((float)vv[j][e] - mx);
          pv[e] = (_Float16)p;
          l += p;
        }
        rp[j] = pv;
      }
      sM[row] = mx; sL[row] = l; sAl[row] = alpha;
    }
    __syncthreads();

    // ---- O = O*alpha + P @ V
#pragma unroll
    for (int ms = 0; ms < 2; ++ms)
#pragma unroll
      for (int ns = 0; ns < 2; ++ns)
#pragma unroll
        for (int r = 0; r < 8; ++r) {
          const int row = wm * 32 + ms * 16 + hs * 8 + r;
          oacc[ms][ns][r] *= sAl[row];
        }
#pragma unroll
    for (int kc = 0; kc < 2; ++kc) {
#pragma unroll
      for (int ms = 0; ms < 2; ++ms) {
        const v16h af = load_a16(&sSP[(wm * 32 + ms * 16) * 64 + kc * 32], 64);
#pragma unroll
        for (int ns = 0; ns < 2; ++ns) {
          const v16h bf = load_b_swz(&sV[(kc * 4 + wn * 2 + ns) * 512]);
          oacc[ms][ns] = wmma_f16(af, bf, oacc[ms][ns]);
        }
      }
    }
    __syncthreads();
  }

  // ---- finalize: divide by l, pack f16 context [B,T,H*D]
#pragma unroll
  for (int ms = 0; ms < 2; ++ms)
#pragma unroll
    for (int ns = 0; ns < 2; ++ns)
#pragma unroll
      for (int r = 0; r < 8; ++r) {
        const int row = wm * 32 + ms * 16 + hs * 8 + r;
        const int col = wn * 32 + ns * 16 + ln;
        const float v = oacc[ms][ns][r] / sL[row];
        const int t = qt0 + row;
        Ctx[((size_t)b * Tsz + t) * HT + h * HD + col] = (_Float16)v;
      }
}

// ===========================================================================
// Kernel 3: out = ctx(f16) @ w_o -> fp32
// ===========================================================================
__global__ void out_proj_kernel(const _Float16* __restrict__ Ctx,
                                const float* __restrict__ Wo,
                                float* __restrict__ Out) {
  __shared__ _Float16 sA[128 * 32];
  __shared__ _Float16 sB[32 * 64];    // swizzled blobs

  const int tid = threadIdx.x;
  const int gr0 = blockIdx.x * 128;
  const int ct0 = blockIdx.y * 64;
  const int wid = tid >> 5, wm = wid & 3, wn = wid >> 2;
  const int lane = tid & 31, hs = lane >> 4, ln = lane & 15;

  const v8f vzero = {};
  v8f acc[2][2];
  acc[0][0] = vzero; acc[0][1] = vzero; acc[1][0] = vzero; acc[1][1] = vzero;

  for (int kc = 0; kc < HT / 32; ++kc) {
    // A tile: straight f16 copy, b128 both sides
#pragma unroll
    for (int i = 0; i < 2; ++i) {
      const int g = tid + i * 256;          // 512 groups of 8 halves
      const int r = g >> 2, c8 = g & 3;
      *(v8h*)&sA[r * 32 + c8 * 8] =
          *(const v8h*)&Ctx[(size_t)(gr0 + r) * HT + kc * 32 + c8 * 8];
    }
    // B tile swizzled
    {
      const int kp = tid >> 4, n4 = tid & 15;
      const int k0 = kp * 2, n0 = n4 * 4;
      const float* wr = &Wo[(size_t)(kc * 32 + k0) * Esz + ct0 + n0];
      const v4f r0 = *(const v4f*)wr;
      const v4f r1 = *(const v4f*)(wr + Esz);
#pragma unroll
      for (int j = 0; j < 4; ++j) {
        const int n   = n0 + j;
        const int off = (n >> 4) * 512 + (((k0 >> 4) << 4) + (n & 15)) * 16 + (k0 & 15);
        *(unsigned*)&sB[off] = pack2(r0[j], r1[j]);
      }
    }
    __syncthreads();
#pragma unroll
    for (int ms = 0; ms < 2; ++ms) {
      const v16h a = load_a16(&sA[(wm * 32 + ms * 16) * 32], 32);
#pragma unroll
      for (int ns = 0; ns < 2; ++ns) {
        const v16h b = load_b_swz(&sB[(wn * 2 + ns) * 512]);
        acc[ms][ns] = wmma_f16(a, b, acc[ms][ns]);
      }
    }
    __syncthreads();
  }

#pragma unroll
  for (int ms = 0; ms < 2; ++ms)
#pragma unroll
    for (int ns = 0; ns < 2; ++ns)
#pragma unroll
      for (int r = 0; r < 8; ++r) {
        const int row = wm * 32 + ms * 16 + hs * 8 + r;
        const int col = wn * 32 + ns * 16 + ln;
        Out[(size_t)(gr0 + row) * Esz + ct0 + col] = acc[ms][ns][r];
      }
}

// ===========================================================================
extern "C" void kernel_launch(void* const* d_in, const int* in_sizes, int n_in,
                              void* d_out, int out_size, void* d_ws, size_t ws_size,
                              hipStream_t stream) {
  (void)in_sizes; (void)n_in; (void)out_size; (void)ws_size;
  const float* Xq  = (const float*)d_in[0];
  const float* Xkv = (const float*)d_in[1];
  const float* Wq  = (const float*)d_in[2];
  const float* Wk  = (const float*)d_in[3];
  const float* Wv  = (const float*)d_in[4];
  const float* Wo  = (const float*)d_in[5];
  float* Out = (float*)d_out;

  const size_t seg = (size_t)Bsz * NH * Tsz * HD;   // 8M halves = 16 MB
  _Float16* Qh  = (_Float16*)d_ws;
  _Float16* Ksw = Qh + seg;
  _Float16* Vsw = Ksw + seg;
  _Float16* Ctx = Vsw + seg;

  qkv_rope_kernel<<<dim3((Bsz * Tsz) / 128, NH, 3), 256, 0, stream>>>(
      Xq, Xkv, Wq, Wk, Wv, Qh, Ksw, Vsw);
  attention_kernel<<<dim3(Tsz / 128, NH, Bsz), 256, 0, stream>>>(
      Qh, Ksw, Vsw, Ctx);
  out_proj_kernel<<<dim3((Bsz * Tsz) / 128, Esz / 64), 256, 0, stream>>>(
      Ctx, Wo, Out);
}
